// MultiHeadAttention_15393162789316
// MI455X (gfx1250) — matrix operand order repaired and checked by reading, not compiled, not run on hardware
//
#include <hip/hip_runtime.h>
#include <hip/hip_bf16.h>
#include <stdint.h>
#include <math.h>

// ---------------------------------------------------------------------------
// MI455X / gfx1250 multi-head attention (faithful to the buggy reference:
// scores = q·v^T, k-projection is dead code and skipped).
//
//   1) convert x_q, x_v to bf16; transpose+convert Wq, Wv, Wf to bf16
//   2) 64x64-per-wave bf16 WMMA GEMMs: Q = x_q@Wq+bq, V = x_v@Wv+bv
//      (V pass also emits a transposed copy Vt[b][h][d][T] nearly for free)
//   3) flash attention, S^T = V·Q^T (D-frag == P A-frag, zero shuffles);
//      per 32-key step the TDM loads BOTH the row-major V tile (score
//      A-frags) and the Vt tile (P·V B-frags) into LDS, double-buffered
//      with s_wait_tensorcnt — all LDS reads are contiguous vector loads
//   4) out = A@Wf + bf  (f32)
// ---------------------------------------------------------------------------

typedef __attribute__((ext_vector_type(16))) __bf16 bf16x16;
typedef __attribute__((ext_vector_type(8)))  __bf16 bf16x8;
typedef __attribute__((ext_vector_type(4)))  __bf16 bf16x4;
typedef __attribute__((ext_vector_type(8)))  float  f32x8;
typedef __attribute__((ext_vector_type(4)))  unsigned int u32x4;
typedef __attribute__((ext_vector_type(4)))  int    i32x4;
typedef __attribute__((ext_vector_type(8)))  int    i32x8;

#define WMMA_BF16(a, b, c) \
  __builtin_amdgcn_wmma_f32_16x16x32_bf16(false, (a), false, (b), (short)0, (c), false, false)

#define DM   1024
#define TT   2048
#define NB   4
#define NH   16
#define DK   64

// ---- fragment loaders -------------------------------------------------------
// A-fragment (16x32 bf16): lane = row M (=lane&15);
// lanes<16 hold K {0..7,16..23}, lanes>=16 hold K {8..15,24..31}.
__device__ __forceinline__ bf16x16 load_a_bf16(const __bf16* tile, int ld, int lane) {
  const __bf16* p = tile + (size_t)(lane & 15) * ld + ((lane >> 4) << 3);
  bf16x8 lo = *(const bf16x8*)p;
  bf16x8 hi = *(const bf16x8*)(p + 16);
  bf16x16 a;
#pragma unroll
  for (int i = 0; i < 8; ++i) { a[i] = lo[i]; a[8 + i] = hi[i]; }
  return a;
}

// B-fragment (32x16 bf16) from a matrix whose rows are B's columns
// (lane = column N = lane&15; lanes<16 K {0..7,16..23}, lanes>=16 K {8..15,24..31}).
__device__ __forceinline__ bf16x16 load_b_bf16t(const __bf16* colBase, int ld, int lane) {
  const __bf16* p = colBase + (size_t)(lane & 15) * ld + ((lane >> 4) << 3);
  bf16x8 lo = *(const bf16x8*)p;
  bf16x8 hi = *(const bf16x8*)(p + 16);
  bf16x16 b;
#pragma unroll
  for (int i = 0; i < 8; ++i) { b[i] = lo[i]; b[8 + i] = hi[i]; }
  return b;
}

// ---- conversion kernels -----------------------------------------------------
__global__ void f32_to_bf16(const float4* __restrict__ src, __bf16* __restrict__ dst, int n4) {
  int i = blockIdx.x * blockDim.x + threadIdx.x;
  if (i >= n4) return;
  float4 v = src[i];
  bf16x4 o;
  o[0] = (__bf16)v.x; o[1] = (__bf16)v.y; o[2] = (__bf16)v.z; o[3] = (__bf16)v.w;
  ((bf16x4*)dst)[i] = o;
}

__global__ void wconvT(const float* __restrict__ W, __bf16* __restrict__ Wt) {
  int idx = blockIdx.x * blockDim.x + threadIdx.x;   // over DM*DM
  if (idx >= DM * DM) return;
  int k = idx >> 10, n = idx & (DM - 1);
  Wt[(size_t)n * DM + k] = (__bf16)W[idx];
}

// ---- 64x64-per-wave bf16 GEMM ----------------------------------------------
// HEADSPLIT=true : Out is bf16 [NB][NH][T][DK]; optional OutT = [NB][NH][DK][T]
// HEADSPLIT=false: Out is f32 row-major [Mtot][DM]
template <bool HEADSPLIT>
__global__ __launch_bounds__(32)
void gemm_bf16(const __bf16* __restrict__ A,    // [.][Mrows][DM] bf16 row-major
               const __bf16* __restrict__ Wt,   // [DM n][DM k] bf16 (transposed)
               const float* __restrict__ bias,  // [DM]
               void* __restrict__ OutP,
               __bf16* __restrict__ OutT,       // transposed copy (or nullptr)
               int Mrows) {
  int lane = threadIdx.x;
  int m0 = blockIdx.x * 64, n0 = blockIdx.y * 64, b = blockIdx.z;
  A += (size_t)b * Mrows * DM;
  f32x8 acc[4][4] = {};
  for (int k0 = 0; k0 < DM; k0 += 32) {
    __builtin_prefetch(A + (size_t)(m0 + (lane & 15)) * DM + k0 + 128, 0, 0);
    __builtin_prefetch(Wt + (size_t)(n0 + (lane & 15)) * DM + k0 + 128, 0, 0);
    bf16x16 af[4], bfr[4];
#pragma unroll
    for (int i = 0; i < 4; ++i)
      af[i] = load_a_bf16(A + (size_t)(m0 + i * 16) * DM + k0, DM, lane);
#pragma unroll
    for (int j = 0; j < 4; ++j)
      bfr[j] = load_b_bf16t(Wt + (size_t)(n0 + j * 16) * DM + k0, DM, lane);
#pragma unroll
    for (int i = 0; i < 4; ++i)
#pragma unroll
      for (int j = 0; j < 4; ++j)
        acc[i][j] = WMMA_BF16(af[i], bfr[j], acc[i][j]);
  }
  // D layout: lane = col N (=lane&15); rows M = (lane>=16?8:0)+r.
#pragma unroll
  for (int i = 0; i < 4; ++i)
#pragma unroll
    for (int j = 0; j < 4; ++j) {
      int col = n0 + j * 16 + (lane & 15);
      float bv = bias[col];
      int mb = m0 + i * 16 + ((lane >> 4) << 3);
      if (HEADSPLIT) {
        int h = col >> 6, d = col & (DK - 1);
        __bf16* p = (__bf16*)OutP + (((size_t)b * NH + h) * TT + mb) * DK + d;
#pragma unroll
        for (int r = 0; r < 8; ++r) p[(size_t)r * DK] = (__bf16)(acc[i][j][r] + bv);
        if (OutT) {  // transposed copy: rows (t) contiguous per lane
          __bf16* p2 = OutT + (((size_t)b * NH + h) * DK + d) * TT + mb;
          bf16x8 o;
#pragma unroll
          for (int r = 0; r < 8; ++r) o[r] = (__bf16)(acc[i][j][r] + bv);
          *(bf16x8*)p2 = o;
        }
      } else {
        float* p = (float*)OutP + (size_t)mb * DM + col;
#pragma unroll
        for (int r = 0; r < 8; ++r) p[(size_t)r * DM] = acc[i][j][r] + bv;
      }
    }
}

// ---- attention: one wave per (batch, head, 16-query block) ------------------
__global__ __launch_bounds__(32)
void attn(const __bf16* __restrict__ Q,    // [NB][NH][T][DK]
          const __bf16* __restrict__ V,    // [NB][NH][T][DK]
          const __bf16* __restrict__ Vt,   // [NB][NH][DK][T]
          const int* __restrict__ mask,    // [NB][T]
          __bf16* __restrict__ Abf) {      // [NB][T][DM]
  __shared__ alignas(64) __bf16 vtileR[2][32 * DK];  // row-major 32 key rows
  __shared__ alignas(64) __bf16 vtileT[2][DK * 32];  // transposed [d][j]
  int lane = threadIdx.x;
  int m0 = blockIdx.x * 16, h = blockIdx.y, b = blockIdx.z;
  const __bf16* Qh  = Q  + ((size_t)b * NH + h) * TT * DK;
  const __bf16* Vh  = V  + ((size_t)b * NH + h) * TT * DK;
  const __bf16* Vth = Vt + ((size_t)b * NH + h) * DK * TT;
  const int* mk = mask + (size_t)b * TT;

  // TDM D# group1 descriptors (CDNA5 ISA bit layout)
  // row-major tile: tensor 64 x 2048 (stride 64), tile 64 x 32, 2B elements
  const i32x8 g1R = { 0x00010000, (int)(64u << 16), (int)(2048u << 16),
                      (int)(64u << 16), 32, 64, 0, 0 };
  // transposed tile: tensor 2048 x 64 (stride 2048), tile 32 x 64, 2B elements
  const i32x8 g1T = { 0x00010000, (int)(2048u << 16), (int)(64u << 16),
                      (int)(32u << 16), 64, 2048, 0, 0 };
  const i32x4 gz4 = { 0, 0, 0, 0 };
  const i32x8 gz8 = { 0, 0, 0, 0, 0, 0, 0, 0 };
  auto tdm_issue = [&](const i32x8& g1, const __bf16* gptr, const __bf16* lptr) {
    unsigned long long ga = (unsigned long long)(uintptr_t)gptr;
    u32x4 g0;
    g0[0] = 1u;                                                 // count=1
    g0[1] = (unsigned)(uintptr_t)lptr;                          // lds_addr
    g0[2] = (unsigned)ga;                                       // global lo
    g0[3] = (unsigned)((ga >> 32) & 0x01FFFFFFu) | 0x80000000u; // hi | type=2
    __builtin_amdgcn_tensor_load_to_lds(g0, g1, gz4, gz4, gz8, 0);
  };

  // B-fragments of Q^T (K = d): column q of Q^T = contiguous row q of Q.
  bf16x16 qb0 = load_b_bf16t(Qh + (size_t)m0 * DK + 0, DK, lane);
  bf16x16 qb1 = load_b_bf16t(Qh + (size_t)m0 * DK + 32, DK, lane);

  f32x8 acc[4] = {};                       // (16 q x 64 d), 4 N-chunks
  float m_run = -3.0e38f, l_run = 0.0f;
  const float scale = 0.125f;              // 1/sqrt(64)

  tdm_issue(g1R, Vh, &vtileR[0][0]);
  tdm_issue(g1T, Vth, &vtileT[0][0]);
  int cur = 0;
  for (int j0 = 0; j0 < TT; j0 += 32) {
    bool more = (j0 + 32) < TT;
    if (more) {                             // overlap next pair with compute
      tdm_issue(g1R, Vh + (size_t)(j0 + 32) * DK, &vtileR[cur ^ 1][0]);
      tdm_issue(g1T, Vth + (j0 + 32), &vtileT[cur ^ 1][0]);
      __builtin_amdgcn_s_wait_tensorcnt(2); // in-order: oldest pair done
    } else {
      __builtin_amdgcn_s_wait_tensorcnt(0);
    }
    const __bf16* vt  = &vtileR[cur][0];
    const __bf16* vtT = &vtileT[cur][0];

    // S^T tiles (rows=j, cols=q): S^T = V_chunk(16x64) x Q^T(64x16)
    f32x8 s[2];
#pragma unroll
    for (int c = 0; c < 2; ++c) {
      bf16x16 va0 = load_a_bf16(vt + c * 16 * DK + 0, DK, lane);
      bf16x16 va1 = load_a_bf16(vt + c * 16 * DK + 32, DK, lane);
      f32x8 z = {};
      z = WMMA_BF16(va0, qb0, z);
      z = WMMA_BF16(va1, qb1, z);
      s[c] = z;
    }

    // scale + mask + per-lane max (lane holds query q=lane&15; j per VGPR)
    float pm = -3.0e38f;
#pragma unroll
    for (int c = 0; c < 2; ++c) {
      int jb = j0 + c * 16 + ((lane >> 4) << 3);
#pragma unroll
      for (int r = 0; r < 8; ++r) {
        float x = s[c][r] * scale;
        if (mk[jb + r] == 0) x = -3.0e38f;
        s[c][r] = x;
        pm = fmaxf(pm, x);
      }
    }
    pm = fmaxf(pm, __shfl_xor(pm, 16));
    float m_new = fmaxf(m_run, pm);
    float alpha = __expf(m_run - m_new);

    // exponentiate -> P A-fragment (S^T D-frag layout == P A-frag layout)
    bf16x16 pA;
    float ps = 0.0f;
#pragma unroll
    for (int c = 0; c < 2; ++c)
#pragma unroll
      for (int r = 0; r < 8; ++r) {
        float e = __expf(s[c][r] - m_new);
        ps += e;
        pA[c * 8 + r] = (__bf16)e;
      }
    l_run = l_run * alpha + ps + __shfl_xor(ps, 16);
    m_run = m_new;

    // per-row alpha broadcast (row r = query ((lane>=16)?8:0)+r, held by lane q)
    float arow[8];
#pragma unroll
    for (int r = 0; r < 8; ++r) arow[r] = __shfl(alpha, ((lane >> 4) << 3) + r, 32);

    // acc = acc*alpha + P(16x32) x V(32x64); B frag contiguous from Vt tile
#pragma unroll
    for (int c = 0; c < 4; ++c) {
#pragma unroll
      for (int r = 0; r < 8; ++r) acc[c][r] *= arow[r];
      const __bf16* vp = vtT + (size_t)(c * 16 + (lane & 15)) * 32 + ((lane >> 4) << 3);
      bf16x8 vlo = *(const bf16x8*)vp;
      bf16x8 vhi = *(const bf16x8*)(vp + 16);
      bf16x16 vb;
#pragma unroll
      for (int i = 0; i < 8; ++i) { vb[i] = vlo[i]; vb[8 + i] = vhi[i]; }
      acc[c] = WMMA_BF16(pA, vb, acc[c]);
    }
    cur ^= 1;
  }

  // finalize: divide each row (query) by softmax denominator, store bf16
  float lrow[8];
#pragma unroll
  for (int r = 0; r < 8; ++r) lrow[r] = __shfl(l_run, ((lane >> 4) << 3) + r, 32);
#pragma unroll
  for (int c = 0; c < 4; ++c) {
    int d = c * 16 + (lane & 15);
    __bf16* p = Abf + ((size_t)b * TT + m0 + ((lane >> 4) << 3)) * DM + h * DK + d;
#pragma unroll
    for (int r = 0; r < 8; ++r) p[(size_t)r * DM] = (__bf16)(acc[c][r] / lrow[r]);
  }
}

// ---- launch -----------------------------------------------------------------
extern "C" void kernel_launch(void* const* d_in, const int* in_sizes, int n_in,
                              void* d_out, int out_size, void* d_ws, size_t ws_size,
                              hipStream_t stream) {
  // inputs: 0 x_k, 1 x_v, 2 x_q, 3 mask, 4 Wk, 5 bk, 6 Wv, 7 bv, 8 Wq, 9 bq, 10 Wf, 11 bf
  const float* x_v = (const float*)d_in[1];
  const float* x_q = (const float*)d_in[2];
  const int*   mk  = (const int*)d_in[3];
  const float* Wv  = (const float*)d_in[6];
  const float* bv  = (const float*)d_in[7];
  const float* Wq  = (const float*)d_in[8];
  const float* bq  = (const float*)d_in[9];
  const float* Wf  = (const float*)d_in[10];
  const float* bf  = (const float*)d_in[11];
  float* out = (float*)d_out;

  char* ws = (char*)d_ws;                                 // ~102 MB used
  __bf16* WvT = (__bf16*)(ws);
  __bf16* WqT = (__bf16*)(ws + ((size_t)2 << 20));
  __bf16* WfT = (__bf16*)(ws + ((size_t)4 << 20));
  __bf16* Xq  = (__bf16*)(ws + ((size_t)6 << 20));        // 16 MB
  __bf16* Xv  = (__bf16*)(ws + ((size_t)22 << 20));       // 16 MB
  __bf16* Qb  = (__bf16*)(ws + ((size_t)38 << 20));       // 16 MB
  __bf16* Vb  = (__bf16*)(ws + ((size_t)54 << 20));       // 16 MB
  __bf16* Ab  = (__bf16*)(ws + ((size_t)70 << 20));       // 16 MB
  __bf16* Vtb = (__bf16*)(ws + ((size_t)86 << 20));       // 16 MB

  const int NX4 = NB * TT * DM / 4;
  f32_to_bf16<<<(NX4 + 255) / 256, 256, 0, stream>>>((const float4*)x_q, Xq, NX4);
  f32_to_bf16<<<(NX4 + 255) / 256, 256, 0, stream>>>((const float4*)x_v, Xv, NX4);

  dim3 cgrid((DM * DM + 255) / 256);
  wconvT<<<cgrid, 256, 0, stream>>>(Wv, WvT);
  wconvT<<<cgrid, 256, 0, stream>>>(Wq, WqT);
  wconvT<<<cgrid, 256, 0, stream>>>(Wf, WfT);

  dim3 pgrid(TT / 64, DM / 64, NB);
  gemm_bf16<true><<<pgrid, 32, 0, stream>>>(Xq, WqT, bq, (void*)Qb, (__bf16*)nullptr, TT);
  gemm_bf16<true><<<pgrid, 32, 0, stream>>>(Xv, WvT, bv, (void*)Vb, Vtb, TT);

  dim3 agrid(TT / 16, NH, NB);
  attn<<<agrid, 32, 0, stream>>>(Qb, Vb, Vtb, mk, Ab);

  dim3 ogrid((NB * TT) / 64, DM / 64, 1);
  gemm_bf16<false><<<ogrid, 32, 0, stream>>>(Ab, WfT, bf, (void*)out, (__bf16*)nullptr, NB * TT);
}
